// GestureRNN_5866925326426
// MI455X (gfx1250) — matrix operand order repaired and checked by reading, not compiled, not run on hardware
//
#include <hip/hip_runtime.h>
#include <cstdint>
#include <cstddef>

#define B_TOTAL    4096
#define T_STEPS    512
#define IN_DIM     10
#define HID        32
#define NCLS       9
#define CT         64                      // timesteps per staged chunk
#define NCHUNK     (T_STEPS / CT)          // 8
#define ROWS       16                      // batch rows per wave (one WMMA N-tile)
#define XFLOATS    (ROWS * CT * IN_DIM)    // 10240 floats per buffer
#define XBYTES     (XFLOATS * 4)           // 40960 B per buffer
#define ROW_STRIDE (T_STEPS * IN_DIM)      // 5120 floats between batch rows of x

typedef _Float16 v16h __attribute__((ext_vector_type(16)));
typedef float    v8f  __attribute__((ext_vector_type(8)));
typedef unsigned v4u  __attribute__((ext_vector_type(4)));
typedef int      v4i  __attribute__((ext_vector_type(4)));
typedef int      v8i  __attribute__((ext_vector_type(8)));

static __device__ __forceinline__ v8f wmma16(v16h a, v16h b, v8f c) {
  // D = A(16x32 f16) * B(32x16 f16) + C(16x16 f32)
  return __builtin_amdgcn_wmma_f32_16x16x32_f16(false, a, false, b, (short)0, c,
                                                false, false);
}

// pack two f32 -> f16x2 (RTZ) and apply ReLU in packed f16 (single v_pk_max, no
// canonicalize; RTZ-then-clamp == clamp-then-RTZ for ReLU).
static __device__ __forceinline__ unsigned pkrelu(float a, float b) {
  auto h = __builtin_amdgcn_cvt_pkrtz(a, b);   // __fp16 x2 -> 32 bits
  unsigned p = __builtin_bit_cast(unsigned, h);
  unsigned r;
  asm("v_pk_max_num_f16 %0, %1, 0" : "=v"(r) : "v"(p));
  return r;
}

// Producer-row permutation: P[i] swaps bits 3 and 4 of the 5-bit row index
// (P = [0..7, 16..23, 8..15, 24..31]). With producing A-matrices row-permuted
// by P, the two D tiles hold hidden {0-7|16-23} and {8-15|24-31} per lane
// group, so packed f16 pairs land directly in B-fragment registers.
static __device__ __forceinline__ int permrow(int i) {
  return (i & 7) | ((i & 8) << 1) | ((i & 16) >> 1);
}

// A-fragment (16x32 f16, layout: M = lane%16, K = j + 16*(lane/16))
// of weight W [rows x cols] row-major; tile m = output rows 16m..16m+15 (optionally
// permuted by P); zero-padded outside [rows x cols].
static __device__ __forceinline__ v16h load_wA(const float* __restrict__ W, int rows,
                                               int cols, int mtile, int lane,
                                               bool perm) {
  v16h a;
  int i = (lane & 15) + 16 * mtile;
  int M = perm ? permrow(i) : i;
  #pragma unroll
  for (int j = 0; j < 16; ++j) {
    int K = j + 16 * (lane >> 4);
    float v = 0.0f;
    if (M < rows && K < cols) v = W[M * cols + K];
    a[j] = (_Float16)v;
  }
  return a;
}

// Bias as C-fragment of the transposed product: C[M=out_idx][N=batch] = bias[out_idx],
// out_idx following the producer's (optionally permuted) row ordering.
// D/C layout: lane holds N = lane%16, M = r + 8*(lane/16).
static __device__ __forceinline__ v8f bias_frag(const float* __restrict__ pa,
                                                const float* __restrict__ pb,
                                                int mtile, int lane, int limit,
                                                bool perm) {
  v8f c;
  #pragma unroll
  for (int r = 0; r < 8; ++r) {
    int i = r + 8 * (lane >> 4) + 16 * mtile;
    int idx = perm ? permrow(i) : i;
    float v = 0.0f;
    if (idx < limit) { v = pa[idx]; if (pb) v += pb[idx]; }
    c[r] = v;
  }
  return c;
}

// Two f32 D tiles (producer rows P-permuted) -> relu'd f16 B fragment
// (32x16, lane holds N = lane%16, K = j + 16*(lane/16)). Thanks to the row
// permutation this is pure per-lane pack+clamp: no cross-lane ops.
static __device__ __forceinline__ v16h pack_relu(v8f t0, v8f t1) {
  union { v16h h; unsigned u[8]; } r;
  #pragma unroll
  for (int q = 0; q < 4; ++q) {
    r.u[q]     = pkrelu(t0[2 * q], t0[2 * q + 1]);
    r.u[4 + q] = pkrelu(t1[2 * q], t1[2 * q + 1]);
  }
  return r.h;
}

// x_t as B fragment: B[k][batch] = x[batch, t, k], K padded 10 -> 32 with zeros.
// Only lanes 0-15 (K 0..15) carry data; K=10..15 and all of lanes 16-31 are zero.
static __device__ __forceinline__ v16h make_bx(const float* xrow, bool low) {
  v16h b;
  #pragma unroll
  for (int j = 0; j < 16; ++j) b[j] = (_Float16)0.0f;
  if (low) {
    #pragma unroll
    for (int j = 0; j < IN_DIM; ++j) b[j] = (_Float16)xrow[j];
  }
  return b;
}

// TDM: DMA a [ROWS x (CT*IN_DIM)] f32 tile (row stride ROW_STRIDE) into LDS.
static __device__ __forceinline__ void tdm_load_tile(unsigned lds_off,
                                                     const float* g) {
  uint64_t ga = (uint64_t)(uintptr_t)g;
  v4u g0;
  g0[0] = 1u;                                              // count=1, user D#
  g0[1] = lds_off;                                         // LDS byte address
  g0[2] = (unsigned)(ga & 0xFFFFFFFFull);                  // global_addr[31:0]
  g0[3] = (unsigned)((ga >> 32) & 0x01FFFFFFull) | 0x80000000u; // addr[56:32] | type=2
  v8i g1;
  g1[0] = 0x00020000;                                      // data_size=4B, no mask
  g1[1] = (int)(((unsigned)(CT * IN_DIM)) << 16);          // tensor_dim0 = 640
  g1[2] = (int)(((unsigned)ROWS) << 16);                   // tensor_dim1 = 16
  g1[3] = (int)(((unsigned)(CT * IN_DIM)) << 16);          // tile_dim0   = 640
  g1[4] = ROWS;                                            // tile_dim1=16, tile_dim2=0
  g1[5] = ROW_STRIDE;                                      // tensor_dim0_stride = 5120
  g1[6] = 0;
  g1[7] = 0;
  v4i z4; z4[0] = 0; z4[1] = 0; z4[2] = 0; z4[3] = 0;
#if defined(__clang_major__) && (__clang_major__ >= 23)
  v8i z8; z8[0]=0; z8[1]=0; z8[2]=0; z8[3]=0; z8[4]=0; z8[5]=0; z8[6]=0; z8[7]=0;
  __builtin_amdgcn_tensor_load_to_lds(g0, g1, z4, z4, z8, 0);
#else
  __builtin_amdgcn_tensor_load_to_lds(g0, g1, z4, z4, 0);
#endif
}

__global__ __launch_bounds__(32) void gesture_rnn_kernel(
    const float* __restrict__ x,
    const float* __restrict__ W_ih0, const float* __restrict__ W_hh0,
    const float* __restrict__ b_ih0, const float* __restrict__ b_hh0,
    const float* __restrict__ W_ih1, const float* __restrict__ W_hh1,
    const float* __restrict__ b_ih1, const float* __restrict__ b_hh1,
    const float* __restrict__ W_fc,  const float* __restrict__ b_fc,
    float* __restrict__ out) {
  extern __shared__ float smem[];                 // 2 x XBYTES x-staging buffers
  const int lane = threadIdx.x & 31;
  const bool low = lane < 16;
  const int b0 = blockIdx.x * ROWS;

  // ---- Weights as A fragments (resident in VGPRs for all 512 steps).
  //      Producers of h1/h2 are row-permuted by P; consumers' K stays true. ----
  v16h a_ih0_0 = load_wA(W_ih0, HID, IN_DIM, 0, lane, true);
  v16h a_ih0_1 = load_wA(W_ih0, HID, IN_DIM, 1, lane, true);
  v16h a_hh0_0 = load_wA(W_hh0, HID, HID, 0, lane, true);
  v16h a_hh0_1 = load_wA(W_hh0, HID, HID, 1, lane, true);
  v16h a_ih1_0 = load_wA(W_ih1, HID, HID, 0, lane, true);
  v16h a_ih1_1 = load_wA(W_ih1, HID, HID, 1, lane, true);
  v16h a_hh1_0 = load_wA(W_hh1, HID, HID, 0, lane, true);
  v16h a_hh1_1 = load_wA(W_hh1, HID, HID, 1, lane, true);
  v16h a_fc    = load_wA(W_fc,  NCLS, HID, 0, lane, false);

  v8f c0_0 = bias_frag(b_ih0, b_hh0, 0, lane, HID, true);
  v8f c0_1 = bias_frag(b_ih0, b_hh0, 1, lane, HID, true);
  v8f c1_0 = bias_frag(b_ih1, b_hh1, 0, lane, HID, true);
  v8f c1_1 = bias_frag(b_ih1, b_hh1, 1, lane, HID, true);
  v8f cfc  = bias_frag(b_fc,  nullptr, 0, lane, NCLS, false);

  v16h bh1, bh2;                                   // hidden state as B fragments
  #pragma unroll
  for (int j = 0; j < 16; ++j) { bh1[j] = (_Float16)0.0f; bh2[j] = (_Float16)0.0f; }

  // ---- Prefetch first two x chunks via TDM (double buffer) ----
  const float* gx = x + (size_t)b0 * ROW_STRIDE;
  tdm_load_tile(0u, gx);
  tdm_load_tile((unsigned)XBYTES, gx + CT * IN_DIM);

  for (int c = 0; c < NCHUNK; ++c) {
    if (c + 1 < NCHUNK) __builtin_amdgcn_s_wait_tensorcnt((short)1);
    else                __builtin_amdgcn_s_wait_tensorcnt((short)0);
    const float* xbuf = smem + (c & 1) * XFLOATS + (lane & 15) * (CT * IN_DIM);

    #pragma unroll 2
    for (int tl = 0; tl < CT; ++tl) {
      // layer 0: H1' = relu(W_ih0 * X_t' + b0 + W_hh0 * H1')   (rows P-permuted)
      v16h bx = make_bx(xbuf + tl * IN_DIM, low);
      v8f acc0 = wmma16(a_ih0_0, bx, c0_0);
      v8f acc1 = wmma16(a_ih0_1, bx, c0_1);
      acc0 = wmma16(a_hh0_0, bh1, acc0);
      acc1 = wmma16(a_hh0_1, bh1, acc1);
      bh1 = pack_relu(acc0, acc1);                 // relu + f16 pack, no shuffles
      // layer 1: H2' = relu(W_ih1 * H1'_new + b1 + W_hh1 * H2'_old)
      v8f d0 = wmma16(a_ih1_0, bh1, c1_0);
      v8f d1 = wmma16(a_ih1_1, bh1, c1_1);
      d0 = wmma16(a_hh1_0, bh2, d0);
      d1 = wmma16(a_hh1_1, bh2, d1);
      bh2 = pack_relu(d0, d1);
    }

    // All LDS reads of this buffer retired before TDM overwrites it.
    asm volatile("s_wait_dscnt 0" ::: "memory");
    if (c + 2 < NCHUNK)
      tdm_load_tile((unsigned)((c & 1) * XBYTES),
                    gx + (size_t)(c + 2) * CT * IN_DIM);
  }

  // ---- FC on last h2: out' = W_fc * H2' + b_fc (one 16x16x32 WMMA) ----
  v8f accf = wmma16(a_fc, bh2, cfc);
  int row = b0 + (lane & 15);
  #pragma unroll
  for (int r = 0; r < 8; ++r) {
    int n = r + 8 * (lane >> 4);
    if (n < NCLS) out[row * NCLS + n] = accf[r];
  }
}

extern "C" void kernel_launch(void* const* d_in, const int* in_sizes, int n_in,
                              void* d_out, int out_size, void* d_ws, size_t ws_size,
                              hipStream_t stream) {
  (void)in_sizes; (void)n_in; (void)out_size; (void)d_ws; (void)ws_size;
  const float* x     = (const float*)d_in[0];
  const float* W_ih0 = (const float*)d_in[1];
  const float* W_hh0 = (const float*)d_in[2];
  const float* b_ih0 = (const float*)d_in[3];
  const float* b_hh0 = (const float*)d_in[4];
  const float* W_ih1 = (const float*)d_in[5];
  const float* W_hh1 = (const float*)d_in[6];
  const float* b_ih1 = (const float*)d_in[7];
  const float* b_hh1 = (const float*)d_in[8];
  const float* W_fc  = (const float*)d_in[9];
  const float* b_fc  = (const float*)d_in[10];
  float* out = (float*)d_out;

  dim3 grid(B_TOTAL / ROWS);   // 256 waves, one 16-row batch tile each
  dim3 block(32);              // one wave32 per workgroup
  gesture_rnn_kernel<<<grid, block, 2 * XBYTES, stream>>>(
      x, W_ih0, W_hh0, b_ih0, b_hh0, W_ih1, W_hh1, b_ih1, b_hh1, W_fc, b_fc, out);
}
